// GAT_71330816852260
// MI455X (gfx1250) — matrix operand order repaired and checked by reading, not compiled, not run on hardware
//
#include <hip/hip_runtime.h>
#include <hip/hip_bf16.h>

#define N_NODES 50000
#define E_EDGES 800000
#define E2      (E_EDGES + N_NODES)   // edges incl. self-loops = 850000
#define F_IN    128
#define HEADS   8
#define HID     16
#define F_MID   128                   // HEADS*HID
#define CLASSES 40
#define NEG_SLOPE 0.2f

typedef __attribute__((ext_vector_type(16))) _Float16 v16h;
typedef __attribute__((ext_vector_type(8)))  float    v8f;

#define GRID(n) (((n) + 255) / 256)

// ---------------- helpers ----------------

__device__ __forceinline__ void edge_sd(const int* __restrict__ ei, int e, int& s, int& d) {
    if (e < E_EDGES) { s = ei[e]; d = ei[E_EDGES + e]; }
    else             { s = e - E_EDGES; d = s; }          // self loop
}

// monotone float -> uint encoding so integer atomicMax == float max
__device__ __forceinline__ unsigned fenc(float f) {
    unsigned u = __float_as_uint(f);
    return (u & 0x80000000u) ? ~u : (u | 0x80000000u);
}
__device__ __forceinline__ float fdec(unsigned u) {
    return (u & 0x80000000u) ? __uint_as_float(u & 0x7fffffffu)
                             : __uint_as_float(~u);
}

__global__ void fill_f32(float* p, float v, int n) {
    int i = blockIdx.x * 256 + threadIdx.x;
    if (i < n) p[i] = v;
}
__global__ void fill_u32(unsigned* p, unsigned v, int n) {
    int i = blockIdx.x * 256 + threadIdx.x;
    if (i < n) p[i] = v;
}

// ---------------- GEMM 1: h1[N,128] = x[N,128] @ W1[128,128] (WMMA f16->f32) ----------
// block = 256 threads = 8 waves; wave w computes the 16x16 tile for cols [16w,16w+16)
// (== head w since HID==16). A-tile staged in LDS as f16.

__global__ __launch_bounds__(256) void gemm1_kernel(const float* __restrict__ x,
                                                    const float* __restrict__ W1,
                                                    float* __restrict__ h1) {
    __shared__ _Float16 xs[16 * F_IN];
    const int row0 = blockIdx.x * 16;
    const int tid  = threadIdx.x;
    for (int i = tid; i < 16 * F_IN; i += 256) {
        int r = i >> 7, cc = i & 127;
        int gr = row0 + r;
        xs[i] = (_Float16)((gr < N_NODES) ? x[gr * F_IN + cc] : 0.f);
    }
    __syncthreads();

    const int wave = tid >> 5;
    const int lane = tid & 31;
    const int half = lane >> 4;     // 0: lanes 0-15, 1: lanes 16-31
    const int l16  = lane & 15;
    const int bcol = wave * 16 + l16;

    v8f acc = {};
#pragma unroll
    for (int kk = 0; kk < 4; ++kk) {
        const int kb = kk * 32;
        v16h a, b;
        const _Float16* xr = xs + l16 * F_IN;   // A row = l16
        // A 16x32 f16 layout: VGPR0-3 = K[half*8 .. +8), VGPR4-7 = K[16+half*8 .. +8)
        const int ak0 = kb + half * 8;
        const int ak1 = kb + 16 + half * 8;
#pragma unroll
        for (int i = 0; i < 8; ++i) a[i]     = xr[ak0 + i];
#pragma unroll
        for (int i = 0; i < 8; ++i) a[8 + i] = xr[ak1 + i];
        // B 32x16 f16 layout: lane col = l16, K = half*8 + (i&7) + 16*(i>>3)
        const int bk = kb + half * 8;
#pragma unroll
        for (int i = 0; i < 16; ++i) b[i] = (_Float16)W1[(bk + (i & 7) + ((i >> 3) << 4)) * F_MID + bcol];
        acc = __builtin_amdgcn_wmma_f32_16x16x32_f16(false, a, false, b,
                                                     (short)0, acc, false, false);
    }
    // C layout: lane col = bcol, VGPR j -> row j + 8*half
#pragma unroll
    for (int j = 0; j < 8; ++j) {
        int gr = row0 + j + 8 * half;
        if (gr < N_NODES) h1[gr * F_MID + bcol] = acc[j];
    }
}

// ---------------- GEMM 2: h2[N,40] = out1[N,128] @ W2[128,40] -------------------------
// block = 128 threads = 4 waves; waves 0..2 cover col tiles 0..2 (cols padded 40->48)

__global__ __launch_bounds__(128) void gemm2_kernel(const float* __restrict__ hin,
                                                    const float* __restrict__ W2,
                                                    float* __restrict__ h2) {
    __shared__ _Float16 xs[16 * F_MID];
    const int row0 = blockIdx.x * 16;
    const int tid  = threadIdx.x;
    for (int i = tid; i < 16 * F_MID; i += 128) {
        int r = i >> 7, cc = i & 127;
        int gr = row0 + r;
        xs[i] = (_Float16)((gr < N_NODES) ? hin[gr * F_MID + cc] : 0.f);
    }
    __syncthreads();

    const int wave = tid >> 5;
    const int lane = tid & 31;
    const int half = lane >> 4;
    const int l16  = lane & 15;
    if (wave >= 3) return;
    const int bcol = wave * 16 + l16;

    v8f acc = {};
#pragma unroll
    for (int kk = 0; kk < 4; ++kk) {
        const int kb = kk * 32;
        v16h a, b;
        const _Float16* xr = xs + l16 * F_MID;
        const int ak0 = kb + half * 8;
        const int ak1 = kb + 16 + half * 8;
#pragma unroll
        for (int i = 0; i < 8; ++i) a[i]     = xr[ak0 + i];
#pragma unroll
        for (int i = 0; i < 8; ++i) a[8 + i] = xr[ak1 + i];
        const int bk = kb + half * 8;
#pragma unroll
        for (int i = 0; i < 16; ++i)
            b[i] = (_Float16)((bcol < CLASSES) ? W2[(bk + (i & 7) + ((i >> 3) << 4)) * CLASSES + bcol] : 0.f);
        acc = __builtin_amdgcn_wmma_f32_16x16x32_f16(false, a, false, b,
                                                     (short)0, acc, false, false);
    }
    if (bcol < CLASSES) {
#pragma unroll
        for (int j = 0; j < 8; ++j) {
            int gr = row0 + j + 8 * half;
            if (gr < N_NODES) h2[gr * CLASSES + bcol] = acc[j];
        }
    }
}

// ---------------- attention coefficients ---------------------------------------------

__global__ void coef1_kernel(const float* __restrict__ h1,
                             const float* __restrict__ a_src, const float* __restrict__ a_dst,
                             float* __restrict__ es, float* __restrict__ ed) {
    int idx = blockIdx.x * 256 + threadIdx.x;           // n*HEADS + h
    if (idx >= N_NODES * HEADS) return;
    int n = idx >> 3, h = idx & 7;
    const float* hp = h1 + n * F_MID + h * HID;
    float s = 0.f, d = 0.f;
#pragma unroll
    for (int c = 0; c < HID; ++c) {
        float v = hp[c];
        s += v * a_src[h * HID + c];
        d += v * a_dst[h * HID + c];
    }
    es[idx] = s; ed[idx] = d;
}

__global__ void coef2_kernel(const float* __restrict__ h2,
                             const float* __restrict__ a_src, const float* __restrict__ a_dst,
                             float* __restrict__ es, float* __restrict__ ed) {
    int n = blockIdx.x * 256 + threadIdx.x;
    if (n >= N_NODES) return;
    const float* hp = h2 + n * CLASSES;
    float s = 0.f, d = 0.f;
#pragma unroll
    for (int c = 0; c < CLASSES; ++c) {
        float v = hp[c];
        s += v * a_src[c];
        d += v * a_dst[c];
    }
    es[n] = s; ed[n] = d;
}

// ---------------- segment softmax: max / exp-sum --------------------------------------

__global__ void edge_max1(const int* __restrict__ ei, const float* __restrict__ es,
                          const float* __restrict__ ed, float* __restrict__ ew,
                          unsigned* __restrict__ amax) {
    int idx = blockIdx.x * 256 + threadIdx.x;           // e*HEADS + h
    if (idx >= E2 * HEADS) return;
    int e = idx >> 3, h = idx & 7;
    int s, d; edge_sd(ei, e, s, d);
    float a = es[s * HEADS + h] + ed[d * HEADS + h];
    a = (a > 0.f) ? a : a * NEG_SLOPE;                  // leaky_relu
    ew[idx] = a;
    atomicMax(&amax[d * HEADS + h], fenc(a));
}

__global__ void edge_sum1(const int* __restrict__ ei, float* __restrict__ ew,
                          const unsigned* __restrict__ amax, float* __restrict__ denom) {
    int idx = blockIdx.x * 256 + threadIdx.x;
    if (idx >= E2 * HEADS) return;
    int e = idx >> 3, h = idx & 7;
    int s, d; edge_sd(ei, e, s, d);
    float w = __expf(ew[idx] - fdec(amax[d * HEADS + h]));
    ew[idx] = w;
    atomicAdd(&denom[d * HEADS + h], w);
}

__global__ void edge_max2(const int* __restrict__ ei, const float* __restrict__ es,
                          const float* __restrict__ ed, float* __restrict__ ew,
                          unsigned* __restrict__ amax) {
    int e = blockIdx.x * 256 + threadIdx.x;
    if (e >= E2) return;
    int s, d; edge_sd(ei, e, s, d);
    float a = es[s] + ed[d];
    a = (a > 0.f) ? a : a * NEG_SLOPE;
    ew[e] = a;
    atomicMax(&amax[d], fenc(a));
}

__global__ void edge_sum2(const int* __restrict__ ei, float* __restrict__ ew,
                          const unsigned* __restrict__ amax, float* __restrict__ denom) {
    int e = blockIdx.x * 256 + threadIdx.x;
    if (e >= E2) return;
    int s, d; edge_sd(ei, e, s, d);
    float w = __expf(ew[e] - fdec(amax[d]));
    ew[e] = w;
    atomicAdd(&denom[d], w);
}

// ---------------- weighted scatter-sum (thread per (edge,feature) for coalescing) -----

__global__ void aggregate1(const int* __restrict__ ei, const float* __restrict__ ew,
                           const float* __restrict__ denom, const float* __restrict__ h1,
                           float* __restrict__ out1) {
    long long idx = (long long)blockIdx.x * 256 + threadIdx.x;   // e*128 + f
    if (idx >= (long long)E2 * F_MID) return;
    int e = (int)(idx >> 7);
    int f = (int)(idx & 127);
    int h = f >> 4;                                   // head = f/HID
    int s, d; edge_sd(ei, e, s, d);
    float coef = ew[e * HEADS + h] / (denom[d * HEADS + h] + 1e-16f);
    atomicAdd(&out1[d * F_MID + f], h1[s * F_MID + f] * coef);
}

__global__ void aggregate2(const int* __restrict__ ei, const float* __restrict__ ew,
                           const float* __restrict__ denom, const float* __restrict__ h2,
                           float* __restrict__ out) {
    long long idx = (long long)blockIdx.x * 256 + threadIdx.x;   // e*40 + c
    if (idx >= (long long)E2 * CLASSES) return;
    int e = (int)(idx / CLASSES);
    int c = (int)(idx - (long long)e * CLASSES);
    int s, d; edge_sd(ei, e, s, d);
    float coef = ew[e] / (denom[d] + 1e-16f);
    atomicAdd(&out[d * CLASSES + c], h2[s * CLASSES + c] * coef);
}

// ---------------- bias / activations --------------------------------------------------

__global__ void bias_elu1(float* __restrict__ out1, const float* __restrict__ b1) {
    int idx = blockIdx.x * 256 + threadIdx.x;
    if (idx >= N_NODES * F_MID) return;
    float v = out1[idx] + b1[idx & 127];
    out1[idx] = (v > 0.f) ? v : (__expf(v) - 1.f);     // ELU
}

__global__ void bias2(float* __restrict__ out, const float* __restrict__ b2) {
    int idx = blockIdx.x * 256 + threadIdx.x;
    if (idx >= N_NODES * CLASSES) return;
    out[idx] += b2[idx % CLASSES];
}

// ---------------- launch --------------------------------------------------------------

extern "C" void kernel_launch(void* const* d_in, const int* in_sizes, int n_in,
                              void* d_out, int out_size, void* d_ws, size_t ws_size,
                              hipStream_t stream) {
    (void)in_sizes; (void)n_in; (void)out_size; (void)ws_size;
    const float* x      = (const float*)d_in[0];
    const int*   ei     = (const int*)  d_in[1];
    const float* W1     = (const float*)d_in[2];
    const float* a_src1 = (const float*)d_in[3];
    const float* a_dst1 = (const float*)d_in[4];
    const float* b1     = (const float*)d_in[5];
    const float* W2     = (const float*)d_in[6];
    const float* a_src2 = (const float*)d_in[7];
    const float* a_dst2 = (const float*)d_in[8];
    const float* b2     = (const float*)d_in[9];
    float* out = (float*)d_out;

    // workspace carve-up (aliased across layers; all stream-ordered)
    char* ws = (char*)d_ws;
    size_t o = 0;
    auto carve = [&](size_t bytes) { void* p = ws + o; o += (bytes + 255) & ~(size_t)255; return p; };
    float*    h1    = (float*)   carve((size_t)N_NODES * F_MID * 4);   // reused as h2
    float*    out1  = (float*)   carve((size_t)N_NODES * F_MID * 4);
    float*    es    = (float*)   carve((size_t)N_NODES * HEADS * 4);   // layer2 aliases [0,N)
    float*    ed    = (float*)   carve((size_t)N_NODES * HEADS * 4);
    unsigned* amax  = (unsigned*)carve((size_t)N_NODES * HEADS * 4);
    float*    denom = (float*)   carve((size_t)N_NODES * HEADS * 4);
    float*    ew    = (float*)   carve((size_t)E2 * HEADS * 4);        // layer2 aliases [0,E2)

    // ---- layer 1 ----
    gemm1_kernel<<<(N_NODES + 15) / 16, 256, 0, stream>>>(x, W1, h1);
    coef1_kernel<<<GRID(N_NODES * HEADS), 256, 0, stream>>>(h1, a_src1, a_dst1, es, ed);
    fill_u32<<<GRID(N_NODES * HEADS), 256, 0, stream>>>(amax, 0u, N_NODES * HEADS);
    fill_f32<<<GRID(N_NODES * HEADS), 256, 0, stream>>>(denom, 0.f, N_NODES * HEADS);
    fill_f32<<<GRID(N_NODES * F_MID), 256, 0, stream>>>(out1, 0.f, N_NODES * F_MID);
    edge_max1<<<GRID(E2 * HEADS), 256, 0, stream>>>(ei, es, ed, ew, amax);
    edge_sum1<<<GRID(E2 * HEADS), 256, 0, stream>>>(ei, ew, amax, denom);
    {
        long long n = (long long)E2 * F_MID;
        aggregate1<<<(unsigned)((n + 255) / 256), 256, 0, stream>>>(ei, ew, denom, h1, out1);
    }
    bias_elu1<<<GRID(N_NODES * F_MID), 256, 0, stream>>>(out1, b1);

    // ---- layer 2 (h2 aliases h1; e/amax/denom/ew buffers reused) ----
    float* h2 = h1;
    gemm2_kernel<<<(N_NODES + 15) / 16, 128, 0, stream>>>(out1, W2, h2);
    coef2_kernel<<<GRID(N_NODES), 256, 0, stream>>>(h2, a_src2, a_dst2, es, ed);
    fill_u32<<<GRID(N_NODES), 256, 0, stream>>>(amax, 0u, N_NODES);
    fill_f32<<<GRID(N_NODES), 256, 0, stream>>>(denom, 0.f, N_NODES);
    fill_f32<<<GRID(N_NODES * CLASSES), 256, 0, stream>>>(out, 0.f, N_NODES * CLASSES);
    edge_max2<<<GRID(E2), 256, 0, stream>>>(ei, es, ed, ew, amax);
    edge_sum2<<<GRID(E2), 256, 0, stream>>>(ei, ew, amax, denom);
    {
        long long n = (long long)E2 * CLASSES;
        aggregate2<<<(unsigned)((n + 255) / 256), 256, 0, stream>>>(ei, ew, denom, h2, out);
    }
    bias2<<<GRID(N_NODES * CLASSES), 256, 0, stream>>>(out, b2);
}
